// CADenseAdd_14534169330228
// MI455X (gfx1250) — compile-verified, matrix operand order checked by reading, not run on hardware
//
#include <hip/hip_runtime.h>

// ---------------------------------------------------------------------------
// Problem sizes (from the reference)
// ---------------------------------------------------------------------------
constexpr int BB    = 2048;  // batch
constexpr int NN    = 4096;  // input features
constexpr int CC    = 1024;  // context features
constexpr int UNITS = 4096;
constexpr int RANK  = 512;

// GEMM tiling
constexpr int BM = 128, BN = 128, BK = 32;
constexpr int THREADS = 256;                 // 8 wave32s
constexpr int LDA_S = BK + 8;                // 40 halves: 16B-aligned, conflict-padded
constexpr int LDB_S = BK + 8;

typedef __attribute__((ext_vector_type(16))) __bf16 v16bf;
typedef __attribute__((ext_vector_type(8)))  float  v8f;

union Frag { uint4 q[2]; v16bf v; };

// ---------------------------------------------------------------------------
// Pre-pass 1: transpose + convert  in[Rows x Cols] f32 -> out[Cols x Rows] bf16
// ---------------------------------------------------------------------------
__global__ __launch_bounds__(256)
void transpose_bf16(const float* __restrict__ in, __bf16* __restrict__ out,
                    int Rows, int Cols) {
  __shared__ float tile[32][33];
  const int c0 = blockIdx.x * 32;
  const int r0 = blockIdx.y * 32;
  const int tx = threadIdx.x;        // 0..31
  const int ty = threadIdx.y;        // 0..7
  #pragma unroll
  for (int i = ty; i < 32; i += 8)
    tile[i][tx] = in[(size_t)(r0 + i) * Cols + c0 + tx];
  __syncthreads();
  #pragma unroll
  for (int i = ty; i < 32; i += 8)
    out[(size_t)(c0 + i) * Rows + r0 + tx] = (__bf16)tile[tx][i];
}

// ---------------------------------------------------------------------------
// Pre-pass 2: elementwise convert f32 -> bf16 (vec4)
// ---------------------------------------------------------------------------
__global__ __launch_bounds__(256)
void convert_bf16(const float* __restrict__ in, __bf16* __restrict__ out,
                  int nvec4) {
  int i = blockIdx.x * 256 + threadIdx.x;
  if (i < nvec4) {
    float4 f = reinterpret_cast<const float4*>(in)[i];
    union { __bf16 h[4]; uint2 q; } u;
    u.h[0] = (__bf16)f.x; u.h[1] = (__bf16)f.y;
    u.h[2] = (__bf16)f.z; u.h[3] = (__bf16)f.w;
    reinterpret_cast<uint2*>(out)[i] = u.q;
  }
}

// ---------------------------------------------------------------------------
// GEMM: Out[M x Nc] = epilogue( A[M x K](bf16) @ Bt[Nc x K](bf16)^T )
// Staging uses CDNA5 async DMA: global_load_async_to_lds_b128 (ASYNCcnt),
// so no VGPRs hold tile data across the WMMA block.
// EPI: 1 -> out(f32)  = acc + vec[n]            (s = S + context @ W)
//      2 -> out(bf16) = acc * vec[m*Nc + n]     (t = xu * s, fed to GEMM3 as A)
//      3 -> out(f32)  = relu(acc + vec[n])      (final bias + relu)
// ---------------------------------------------------------------------------
template<int EPI>
__global__ __launch_bounds__(THREADS)
void gemm_bf16_wmma(const __bf16* __restrict__ A, const __bf16* __restrict__ Bt,
                    void* __restrict__ OutP, const float* __restrict__ vec,
                    int M, int Nc, int K) {
  __shared__ __bf16 As[2][BM * LDA_S];
  __shared__ __bf16 Bs[2][BN * LDB_S];        // stored [n][k]

  const int tid  = threadIdx.x;
  const int m0   = blockIdx.y * BM;
  const int n0   = blockIdx.x * BN;
  const int wave = tid >> 5;
  const int lane = tid & 31;
  const int wm   = wave & 3;    // 0..3 : 32-row strip
  const int wn   = wave >> 2;   // 0..1 : 64-col strip
  const int lm   = lane & 15;   // row (A) / col (B) within 16
  const int hi   = lane >> 4;   // K-octet select per ISA layout

  v8f acc[2][4];
  #pragma unroll
  for (int i = 0; i < 2; ++i)
    #pragma unroll
    for (int j = 0; j < 4; ++j)
      acc[i][j] = (v8f)(0.0f);

  // Issue async DMA for one BK-tile into LDS buffer `buf` (16B per lane-op).
  auto issue_tile = [&](int kt, int buf) {
    #pragma unroll
    for (int v = 0; v < 2; ++v) {
      int e = tid + v * THREADS;            // 0..511, 8 bf16 per op
      int r = e >> 2;                       // BK/8 = 4 octets per row
      int k = (e & 3) << 3;
      const __bf16* ga = A  + (size_t)(m0 + r) * K + kt + k;
      const __bf16* gb = Bt + (size_t)(n0 + r) * K + kt + k;
      unsigned la = (unsigned)(size_t)(&As[buf][r * LDA_S + k]);
      unsigned lb = (unsigned)(size_t)(&Bs[buf][r * LDB_S + k]);
      asm volatile("global_load_async_to_lds_b128 %0, %1, off"
                   :: "v"(la), "v"(ga) : "memory");
      asm volatile("global_load_async_to_lds_b128 %0, %1, off"
                   :: "v"(lb), "v"(gb) : "memory");
    }
  };

  // Pipeline prologue: DMA tile 0 into LDS buffer 0
  issue_tile(0, 0);
  int buf = 0;

  for (int kt = 0; kt < K; kt += BK) {
    asm volatile("s_wait_asynccnt 0x0" ::: "memory");  // this wave's DMAs done
    __syncthreads();                    // all waves' DMAs done; prev reads done

    if (kt + BK < K) issue_tile(kt + BK, buf ^ 1);  // DMA next tile while we
                                                    // compute from this one

    Frag fa[2];
    #pragma unroll
    for (int i = 0; i < 2; ++i) {
      const __bf16* ap = &As[buf][(wm * 32 + i * 16 + lm) * LDA_S];
      fa[i].q[0] = *reinterpret_cast<const uint4*>(ap + hi * 8);
      fa[i].q[1] = *reinterpret_cast<const uint4*>(ap + 16 + hi * 8);
    }
    #pragma unroll
    for (int j = 0; j < 4; ++j) {
      Frag fb;
      const __bf16* bp = &Bs[buf][(wn * 64 + j * 16 + lm) * LDB_S];
      fb.q[0] = *reinterpret_cast<const uint4*>(bp + hi * 8);
      fb.q[1] = *reinterpret_cast<const uint4*>(bp + 16 + hi * 8);
      #pragma unroll
      for (int i = 0; i < 2; ++i) {
        acc[i][j] = __builtin_amdgcn_wmma_f32_16x16x32_bf16(
            false, fa[i].v, false, fb.v, (short)0, acc[i][j], false, false);
      }
    }
    buf ^= 1;
  }

  // ---- epilogue: 16x16 f32 C/D layout: m = hi*8 + r, n = lane%16 ----------
  #pragma unroll
  for (int i = 0; i < 2; ++i) {
    #pragma unroll
    for (int j = 0; j < 4; ++j) {
      #pragma unroll
      for (int r = 0; r < 8; ++r) {
        int m = m0 + wm * 32 + i * 16 + hi * 8 + r;
        int n = n0 + wn * 64 + j * 16 + lm;
        float val = acc[i][j][r];
        if (EPI == 1) {
          val += vec[n];
          ((float*)OutP)[(size_t)m * Nc + n] = val;
        } else if (EPI == 2) {
          val *= vec[(size_t)m * Nc + n];
          ((__bf16*)OutP)[(size_t)m * Nc + n] = (__bf16)val;
        } else {
          val += vec[n];
          ((float*)OutP)[(size_t)m * Nc + n] = fmaxf(val, 0.0f);
        }
      }
    }
  }
}

// ---------------------------------------------------------------------------
// Host launcher
// ---------------------------------------------------------------------------
extern "C" void kernel_launch(void* const* d_in, const int* in_sizes, int n_in,
                              void* d_out, int out_size, void* d_ws, size_t ws_size,
                              hipStream_t stream) {
  (void)in_sizes; (void)n_in; (void)out_size; (void)ws_size;
  const float* inputs  = (const float*)d_in[0];   // [B, N]
  const float* context = (const float*)d_in[1];   // [B, C]
  const float* U       = (const float*)d_in[2];   // [N, R]
  const float* S       = (const float*)d_in[3];   // [R]
  const float* V       = (const float*)d_in[4];   // [UNITS, R]
  const float* W       = (const float*)d_in[5];   // [C, R]
  const float* bias    = (const float*)d_in[6];   // [UNITS]
  float* out = (float*)d_out;                     // [B, UNITS]

  // Workspace layout
  char* ws = (char*)d_ws;
  float*  s_buf = (float*)ws;   ws += (size_t)BB * RANK * 4;   // 4 MB (f32)
  __bf16* t_buf = (__bf16*)ws;  ws += (size_t)BB * RANK * 2;   // 2 MB (bf16)
  __bf16* Xb    = (__bf16*)ws;  ws += (size_t)BB * NN * 2;     // 16 MB
  __bf16* Cb    = (__bf16*)ws;  ws += (size_t)BB * CC * 2;     // 4 MB
  __bf16* Ut    = (__bf16*)ws;  ws += (size_t)RANK * NN * 2;   // 4 MB
  __bf16* Wt    = (__bf16*)ws;  ws += (size_t)RANK * CC * 2;   // 1 MB
  __bf16* Vb    = (__bf16*)ws;                                 // 4 MB

  // --- one-time bf16 prep (read-once; L2-resident afterwards) --------------
  convert_bf16<<<(BB * NN / 4 + 255) / 256, 256, 0, stream>>>(
      inputs, Xb, BB * NN / 4);                // Xb[B x N]
  convert_bf16<<<(BB * CC / 4 + 255) / 256, 256, 0, stream>>>(
      context, Cb, BB * CC / 4);               // Cb[B x C]
  convert_bf16<<<(UNITS * RANK / 4 + 255) / 256, 256, 0, stream>>>(
      V, Vb, UNITS * RANK / 4);                // Vb[UNITS x R] (rows run along K)
  transpose_bf16<<<dim3(RANK / 32, CC / 32), dim3(32, 8), 0, stream>>>(
      W, Wt, CC, RANK);                        // Wt[R x C]
  transpose_bf16<<<dim3(RANK / 32, NN / 32), dim3(32, 8), 0, stream>>>(
      U, Ut, NN, RANK);                        // Ut[R x N]

  dim3 blk(THREADS);
  // 1) s = S + context @ W          M=B, N=RANK, K=C     (f32 out)
  gemm_bf16_wmma<1><<<dim3(RANK / BN, BB / BM), blk, 0, stream>>>(
      Cb, Wt, s_buf, S, BB, RANK, CC);
  // 2) t = (inputs @ U) * s         M=B, N=RANK, K=N     (bf16 out)
  gemm_bf16_wmma<2><<<dim3(RANK / BN, BB / BM), blk, 0, stream>>>(
      Xb, Ut, t_buf, s_buf, BB, RANK, NN);
  // 3) out = relu(t @ V^T + bias)   M=B, N=UNITS, K=RANK (f32 out)
  gemm_bf16_wmma<3><<<dim3(UNITS / BN, BB / BM), blk, 0, stream>>>(
      t_buf, Vb, out, bias, BB, UNITS, RANK);
}
